// GatedLinearAttention2_48541720379570
// MI455X (gfx1250) — matrix operand order repaired and checked by reading, not compiled, not run on hardware
//
#include <hip/hip_runtime.h>
#include <hip/hip_bf16.h>
#include <stdint.h>

typedef __attribute__((ext_vector_type(16))) __bf16 v16bf;
typedef __attribute__((ext_vector_type(8)))  float  v8f;
typedef unsigned short u16;

// ---------- helpers ----------
__device__ __forceinline__ u16 f2us(float f) {           // f32 -> bf16 bits, RNE
  unsigned u = __builtin_bit_cast(unsigned, f);
  unsigned r = u + 0x7FFFu + ((u >> 16) & 1u);
  return (u16)(r >> 16);
}
__device__ __forceinline__ __bf16 usbf(u16 s) { return __builtin_bit_cast(__bf16, s); }

__device__ __forceinline__ v8f wmma_bf16(v16bf a, v16bf b, v8f c) {
  // D = A(16x32) * B(32x16) + C, f32 accum
  return __builtin_amdgcn_wmma_f32_16x16x32_bf16(false, a, false, b, (short)0, c, false, false);
}

// ---- async global->LDS (gfx1250 GLOBAL_LOAD_ASYNC_TO_LDS_B128, ASYNCcnt) ----
#if __has_builtin(__builtin_amdgcn_global_load_async_to_lds_b128)
#define USE_ASYNC_LDS 1
#endif

__device__ __forceinline__ void async_copy16(u16* lds, const u16* g) {
#ifdef USE_ASYNC_LDS
  typedef int v4i_ __attribute__((vector_size(16)));
  __builtin_amdgcn_global_load_async_to_lds_b128((v4i_*)g, (v4i_*)lds, 0, 0);
#else
  *(uint4*)lds = *(const uint4*)g;
#endif
}

template <int N>
__device__ __forceinline__ void wait_async() {
#if __has_builtin(__builtin_amdgcn_s_wait_asynccnt)
  __builtin_amdgcn_s_wait_asynccnt(N);
#else
  asm volatile("s_wait_asynccnt %0" ::"n"(N) : "memory");
#endif
}

// cooperative async copy of rows x 32 bf16 tile into LDS (256 threads)
__device__ __forceinline__ void load_tile_async(u16* lds, const u16* g, int ld, int rows) {
  for (int i = threadIdx.x; i < rows * 4; i += 256) {
    int r = i >> 2, c8 = (i & 3) * 8;
    async_copy16(lds + r * 32 + c8, g + (size_t)r * ld + c8);
  }
}

// ---- WMMA fragment builders (contiguous k order: element s of lane half h holds k = 16h + s) ----
// A fragment, logical A[m][k] = s[(m0+m)*ld + k0+k]   (16x32 bf16)
__device__ __forceinline__ v16bf frag_a_rm(const u16* s, int ld, int m0, int k0) {
  int lane = threadIdx.x & 31;
  int m = m0 + (lane & 15);
  int h = lane >> 4;
  v16bf a;
#pragma unroll
  for (int j = 0; j < 8; ++j) {
    int kk = k0 + (h << 4) + (j << 1);
    a[2 * j]     = usbf(s[(size_t)m * ld + kk]);
    a[2 * j + 1] = usbf(s[(size_t)m * ld + kk + 1]);
  }
  return a;
}
// A fragment from transposed storage: logical A[m][k] = s[(k0+k)*ld + m0+m]
__device__ __forceinline__ v16bf frag_a_cm(const u16* s, int ld, int m0, int k0) {
  int lane = threadIdx.x & 31;
  int m = m0 + (lane & 15);
  int h = lane >> 4;
  v16bf a;
#pragma unroll
  for (int j = 0; j < 8; ++j) {
    int kk = k0 + (h << 4) + (j << 1);
    a[2 * j]     = usbf(s[(size_t)kk * ld + m]);
    a[2 * j + 1] = usbf(s[(size_t)(kk + 1) * ld + m]);
  }
  return a;
}
// B fragment, weight-style storage: B[k][n] = s[(n0+n)*ld + k0+k]   (32x16)
__device__ __forceinline__ v16bf frag_b_nk(const u16* s, int ld, int n0, int k0) {
  int lane = threadIdx.x & 31;
  int n = n0 + (lane & 15);
  int h = lane >> 4;
  v16bf b;
#pragma unroll
  for (int j = 0; j < 8; ++j) {
    int kk = k0 + (h << 4) + (j << 1);
    b[2 * j]     = usbf(s[(size_t)n * ld + kk]);
    b[2 * j + 1] = usbf(s[(size_t)n * ld + kk + 1]);
  }
  return b;
}
// B fragment, natural storage: B[k][n] = s[(k0+k)*ld + n0+n]
__device__ __forceinline__ v16bf frag_b_kn(const u16* s, int ld, int n0, int k0) {
  int lane = threadIdx.x & 31;
  int n = n0 + (lane & 15);
  int h = lane >> 4;
  v16bf b;
#pragma unroll
  for (int j = 0; j < 8; ++j) {
    int kk = k0 + (h << 4) + (j << 1);
    b[2 * j]     = usbf(s[(size_t)kk * ld + n]);
    b[2 * j + 1] = usbf(s[(size_t)(kk + 1) * ld + n]);
  }
  return b;
}
// B fragment from f32 storage (converted on the fly): B[k][n] = s[(k0+k)*ld + n0+n]
__device__ __forceinline__ v16bf frag_b_kn_f32(const float* s, int ld, int n0, int k0) {
  int lane = threadIdx.x & 31;
  int n = n0 + (lane & 15);
  int h = lane >> 4;
  v16bf b;
#pragma unroll
  for (int j = 0; j < 8; ++j) {
    int kk = k0 + (h << 4) + (j << 1);
    b[2 * j]     = usbf(f2us(s[(size_t)kk * ld + n]));
    b[2 * j + 1] = usbf(f2us(s[(size_t)(kk + 1) * ld + n]));
  }
  return b;
}
// store 16x16 f32 tile (dst pre-offset to tile origin)
__device__ __forceinline__ void store_f32(float* dst, int ld, v8f c) {
  int lane = threadIdx.x & 31, nn = lane & 15, h = lane >> 4;
#pragma unroll
  for (int r = 0; r < 8; ++r) dst[(size_t)(r + 8 * h) * ld + nn] = c[r];
}
__device__ __forceinline__ void store_bf16(u16* dst, int ld, v8f c) {
  int lane = threadIdx.x & 31, nn = lane & 15, h = lane >> 4;
#pragma unroll
  for (int r = 0; r < 8; ++r) dst[(size_t)(r + 8 * h) * ld + nn] = f2us(c[r]);
}

// ---------- kernels ----------
__global__ __launch_bounds__(256) void cvt_bf16_kernel(const float* src, u16* dst, int n) {
  for (int i = blockIdx.x * 256 + threadIdx.x; i < n; i += gridDim.x * 256)
    dst[i] = f2us(src[i]);
}

// sq/sk = x @ Wq^T , x @ Wk^T    (M=8192, N=128, K=2048), async double-buffered
__global__ __launch_bounds__(256) void qk_proj_kernel(const u16* xb, const u16* wqb,
                                                      const u16* wkb, float* sq, float* sk) {
  __shared__ u16 xs[2][128 * 32], wqs[2][128 * 32], wks[2][128 * 32];
  int m0 = blockIdx.x * 128;
  int wave = threadIdx.x >> 5, wm = wave & 3, wn = wave >> 2;  // 4(M) x 2(N), wave tile 32x64
  const u16* gX = xb + (size_t)m0 * 2048;
  v8f aq[2][4] = {}, ak[2][4] = {};
  load_tile_async(xs[0], gX, 2048, 128);
  load_tile_async(wqs[0], wqb, 2048, 128);
  load_tile_async(wks[0], wkb, 2048, 128);
  for (int it = 0; it < 64; ++it) {
    int cur = it & 1;
    if (it + 1 < 64) {
      int nk = (it + 1) * 32;
      load_tile_async(xs[1 - cur], gX + nk, 2048, 128);
      load_tile_async(wqs[1 - cur], wqb + nk, 2048, 128);
      load_tile_async(wks[1 - cur], wkb + nk, 2048, 128);
      wait_async<6>();
    } else {
      wait_async<0>();
    }
    __syncthreads();
    v16bf a[2], bq[4], bk[4];
#pragma unroll
    for (int mi = 0; mi < 2; ++mi) a[mi] = frag_a_rm(xs[cur], 32, wm * 32 + mi * 16, 0);
#pragma unroll
    for (int ni = 0; ni < 4; ++ni) {
      bq[ni] = frag_b_nk(wqs[cur], 32, wn * 64 + ni * 16, 0);
      bk[ni] = frag_b_nk(wks[cur], 32, wn * 64 + ni * 16, 0);
    }
#pragma unroll
    for (int mi = 0; mi < 2; ++mi)
#pragma unroll
      for (int ni = 0; ni < 4; ++ni) {
        aq[mi][ni] = wmma_bf16(a[mi], bq[ni], aq[mi][ni]);
        ak[mi][ni] = wmma_bf16(a[mi], bk[ni], ak[mi][ni]);
      }
    __syncthreads();
  }
#pragma unroll
  for (int mi = 0; mi < 2; ++mi)
#pragma unroll
    for (int ni = 0; ni < 4; ++ni) {
      size_t o = (size_t)(m0 + wm * 32 + mi * 16) * 128 + wn * 64 + ni * 16;
      store_f32(sq + o, 128, aq[mi][ni]);
      store_f32(sk + o, 128, ak[mi][ni]);
    }
}

// q = softmax(sq) over 128; k = sigmoid(sk). One wave per row.
__global__ __launch_bounds__(256) void act_kernel(float* sq, float* sk) {
  int row = blockIdx.x * 8 + (threadIdx.x >> 5);
  int lane = threadIdx.x & 31;
  float* q = sq + (size_t)row * 128;
  float* k = sk + (size_t)row * 128;
  float v[4], mx = -1e30f;
#pragma unroll
  for (int i = 0; i < 4; ++i) { v[i] = q[lane + 32 * i]; mx = fmaxf(mx, v[i]); }
#pragma unroll
  for (int o = 16; o; o >>= 1) mx = fmaxf(mx, __shfl_xor(mx, o, 32));
  float s = 0.f;
#pragma unroll
  for (int i = 0; i < 4; ++i) { v[i] = __expf(v[i] - mx); s += v[i]; }
#pragma unroll
  for (int o = 16; o; o >>= 1) s += __shfl_xor(s, o, 32);
  float inv = 1.f / s;
#pragma unroll
  for (int i = 0; i < 4; ++i) q[lane + 32 * i] = v[i] * inv;
#pragma unroll
  for (int i = 0; i < 4; ++i) {
    float kk = k[lane + 32 * i];
    k[lane + 32 * i] = 1.f / (1.f + __expf(-kk));
  }
}

// per (batch,chunk): gates, cumsum, intra matrix A, qg, kd, decay
__global__ __launch_bounds__(256) void chunk_prep_kernel(const float* sq, const float* sk,
                                                         u16* Ab, u16* qgb, u16* kdb,
                                                         float* decay) {
  __shared__ float ks[64 * 128];
  __shared__ float Bs[64 * 128];
  int bc = blockIdx.x;          // 0..127  (batch*64 + chunk); rows contiguous
  int row0 = bc * 64;
  int tid = threadIdx.x;
  for (int i = tid; i < 64 * 128; i += 256) ks[i] = sk[(size_t)row0 * 128 + i];
  __syncthreads();
  if (tid < 128) {              // cumsum of gf = log_sigmoid(-k) = -log1p(exp(k)) per column
    float run = 0.f;
    for (int i = 0; i < 64; ++i) {
      run += -log1pf(__expf(ks[i * 128 + tid]));
      Bs[i * 128 + tid] = run;
    }
  }
  __syncthreads();
  for (int e = tid; e < 64 * 64; e += 256) {   // A[i][j], lower-triangular
    int i = e >> 6, j = e & 63;
    float a = 0.f;
    if (j <= i) {
      const float* qi = sq + ((size_t)row0 + i) * 128;
      for (int d = 0; d < 128; ++d)
        a += qi[d] * ks[j * 128 + d] * __expf(Bs[i * 128 + d] - Bs[j * 128 + d]);
    }
    Ab[(size_t)bc * 4096 + e] = f2us(a);
  }
  for (int e = tid; e < 64 * 128; e += 256) {
    int i = e >> 7, d = e & 127;
    float Bl = Bs[63 * 128 + d], Bi = Bs[i * 128 + d];
    float qv = sq[((size_t)row0 + i) * 128 + d];
    qgb[(size_t)row0 * 128 + e] = f2us(qv * __expf(Bi));
    kdb[(size_t)row0 * 128 + e] = f2us(ks[e] * __expf(Bl - Bi));
    if (i == 0) decay[(size_t)bc * 128 + d] = __expf(Bl);
  }
}

// sequential chunk scan per (batch, 64-wide dv tile); S (128x64 f32) lives in LDS
__global__ __launch_bounds__(256) void scan_kernel(const u16* xb, const u16* Ab,
                                                   const u16* qgb, const u16* kdb,
                                                   const float* decay, float* o_f, u16* o_b) {
  __shared__ float S[128 * 64];
  int b = blockIdx.x >> 5;
  int v0 = (blockIdx.x & 31) * 64;
  int tid = threadIdx.x;
  for (int i = tid; i < 128 * 64; i += 256) S[i] = 0.f;
  __syncthreads();
  int wave = tid >> 5;
  int wmO = wave & 3, wnO = wave >> 2;   // o: 4(M16) x 2(N32)
  int dS0 = wave * 16;                   // S-update: wave owns 16 d-rows
  for (int c = 0; c < 64; ++c) {
    int row0 = b * 4096 + c * 64;
    const u16* vh = xb + (size_t)row0 * 2048 + v0;           // [j][n] ld 2048
    const u16* Ac = Ab + (size_t)(b * 64 + c) * 4096;        // 64x64
    const u16* qg = qgb + (size_t)row0 * 128;                // 64x128
    const u16* kd = kdb + (size_t)row0 * 128;                // 64x128
    const float* dc = decay + (size_t)(b * 64 + c) * 128;

    // o = A @ vh + qg @ S
    v8f acc[2] = {};
#pragma unroll
    for (int k0 = 0; k0 < 64; k0 += 32) {
      v16bf a = frag_a_rm(Ac, 64, wmO * 16, k0);
#pragma unroll
      for (int ni = 0; ni < 2; ++ni)
        acc[ni] = wmma_bf16(a, frag_b_kn(vh, 2048, wnO * 32 + ni * 16, k0), acc[ni]);
    }
#pragma unroll
    for (int k0 = 0; k0 < 128; k0 += 32) {
      v16bf a = frag_a_rm(qg, 128, wmO * 16, k0);
#pragma unroll
      for (int ni = 0; ni < 2; ++ni)
        acc[ni] = wmma_bf16(a, frag_b_kn_f32(S, 64, wnO * 32 + ni * 16, k0), acc[ni]);
    }
#pragma unroll
    for (int ni = 0; ni < 2; ++ni) {
      size_t o = (size_t)(row0 + wmO * 16) * 2048 + v0 + wnO * 32 + ni * 16;
      store_f32(o_f + o, 2048, acc[ni]);
      store_bf16(o_b + o, 2048, acc[ni]);
    }

    // S = diag(decay) * S + kd^T @ vh
    v8f accS[4] = {};
#pragma unroll
    for (int k0 = 0; k0 < 64; k0 += 32) {
      v16bf a = frag_a_cm(kd, 128, dS0, k0);   // A[d][j] = kd[j][d]
#pragma unroll
      for (int ni = 0; ni < 4; ++ni)
        accS[ni] = wmma_bf16(a, frag_b_kn(vh, 2048, ni * 16, k0), accS[ni]);
    }
    __syncthreads();                            // all reads of S above are done
    {
      int lane = tid & 31, nn = lane & 15, h = lane >> 4;
#pragma unroll
      for (int ni = 0; ni < 4; ++ni)
#pragma unroll
        for (int r = 0; r < 8; ++r) {
          int d = dS0 + r + 8 * h, n = ni * 16 + nn;
          S[d * 64 + n] = dc[d] * S[d * 64 + n] + accS[ni][r];
        }
    }
    __syncthreads();
  }
}

__global__ __launch_bounds__(256) void rnorm_kernel(const float* o_f, float* rn) {
  int row = blockIdx.x * 8 + (threadIdx.x >> 5);
  int lane = threadIdx.x & 31;
  const float* o = o_f + (size_t)row * 2048;
  float s = 0.f;
  for (int i = lane; i < 2048; i += 32) { float v = o[i]; s += v * v; }
#pragma unroll
  for (int off = 16; off; off >>= 1) s += __shfl_xor(s, off, 32);
  if (lane == 0) rn[row] = rsqrtf(s * (1.f / 2048.f) + 1e-5f);
}

// go = o@Wog^T + x@Wig^T; gated = rmsnorm(o)*gnw * silu(go)  -> bf16
// async double-buffered, dual-WMMA K loop
__global__ __launch_bounds__(256) void gate_kernel(const u16* ob, const u16* xb,
                                                   const u16* wog, const u16* wig,
                                                   const float* o_f, const float* rn,
                                                   const float* gnw, u16* gated) {
  __shared__ u16 os[2][128 * 32], xs[2][128 * 32], wgs[2][128 * 32], wis[2][128 * 32];
  int m0 = blockIdx.x * 128, n0 = blockIdx.y * 128;
  int wave = threadIdx.x >> 5, wm = wave & 3, wn = wave >> 2;
  const u16* gO = ob + (size_t)m0 * 2048;
  const u16* gX = xb + (size_t)m0 * 2048;
  const u16* gG = wog + (size_t)n0 * 2048;
  const u16* gI = wig + (size_t)n0 * 2048;
  v8f acc[2][4] = {};
  load_tile_async(os[0], gO, 2048, 128);
  load_tile_async(xs[0], gX, 2048, 128);
  load_tile_async(wgs[0], gG, 2048, 128);
  load_tile_async(wis[0], gI, 2048, 128);
  for (int it = 0; it < 64; ++it) {
    int cur = it & 1;
    if (it + 1 < 64) {
      int nk = (it + 1) * 32;
      load_tile_async(os[1 - cur], gO + nk, 2048, 128);
      load_tile_async(xs[1 - cur], gX + nk, 2048, 128);
      load_tile_async(wgs[1 - cur], gG + nk, 2048, 128);
      load_tile_async(wis[1 - cur], gI + nk, 2048, 128);
      wait_async<8>();
    } else {
      wait_async<0>();
    }
    __syncthreads();
    v16bf ao[2], ax[2], bg[4], bi[4];
#pragma unroll
    for (int mi = 0; mi < 2; ++mi) {
      ao[mi] = frag_a_rm(os[cur], 32, wm * 32 + mi * 16, 0);
      ax[mi] = frag_a_rm(xs[cur], 32, wm * 32 + mi * 16, 0);
    }
#pragma unroll
    for (int ni = 0; ni < 4; ++ni) {
      bg[ni] = frag_b_nk(wgs[cur], 32, wn * 64 + ni * 16, 0);
      bi[ni] = frag_b_nk(wis[cur], 32, wn * 64 + ni * 16, 0);
    }
#pragma unroll
    for (int mi = 0; mi < 2; ++mi)
#pragma unroll
      for (int ni = 0; ni < 4; ++ni) {
        acc[mi][ni] = wmma_bf16(ao[mi], bg[ni], acc[mi][ni]);
        acc[mi][ni] = wmma_bf16(ax[mi], bi[ni], acc[mi][ni]);
      }
    __syncthreads();
  }
  int lane = threadIdx.x & 31, nn = lane & 15, h = lane >> 4;
#pragma unroll
  for (int mi = 0; mi < 2; ++mi)
#pragma unroll
    for (int ni = 0; ni < 4; ++ni)
#pragma unroll
      for (int r = 0; r < 8; ++r) {
        int m = m0 + wm * 32 + mi * 16 + r + 8 * h;
        int n = n0 + wn * 64 + ni * 16 + nn;
        float go = acc[mi][ni][r];
        float ov = o_f[(size_t)m * 2048 + n];
        float on = ov * rn[m] * gnw[n];
        float sig = 1.f / (1.f + __expf(-go));
        gated[(size_t)m * 2048 + n] = f2us(on * go * sig);
      }
}

// out = gated @ Wo^T  (f32), async double-buffered
__global__ __launch_bounds__(256) void out_proj_kernel(const u16* gb, const u16* wob, float* out) {
  __shared__ u16 as_[2][128 * 32], bs_[2][128 * 32];
  int m0 = blockIdx.x * 128, n0 = blockIdx.y * 128;
  int wave = threadIdx.x >> 5, wm = wave & 3, wn = wave >> 2;
  const u16* gA = gb + (size_t)m0 * 2048;
  const u16* gB = wob + (size_t)n0 * 2048;
  v8f acc[2][4] = {};
  load_tile_async(as_[0], gA, 2048, 128);
  load_tile_async(bs_[0], gB, 2048, 128);
  for (int it = 0; it < 64; ++it) {
    int cur = it & 1;
    if (it + 1 < 64) {
      int nk = (it + 1) * 32;
      load_tile_async(as_[1 - cur], gA + nk, 2048, 128);
      load_tile_async(bs_[1 - cur], gB + nk, 2048, 128);
      wait_async<4>();
    } else {
      wait_async<0>();
    }
    __syncthreads();
    v16bf a[2], bfr[4];
#pragma unroll
    for (int mi = 0; mi < 2; ++mi) a[mi] = frag_a_rm(as_[cur], 32, wm * 32 + mi * 16, 0);
#pragma unroll
    for (int ni = 0; ni < 4; ++ni) bfr[ni] = frag_b_nk(bs_[cur], 32, wn * 64 + ni * 16, 0);
#pragma unroll
    for (int mi = 0; mi < 2; ++mi)
#pragma unroll
      for (int ni = 0; ni < 4; ++ni) acc[mi][ni] = wmma_bf16(a[mi], bfr[ni], acc[mi][ni]);
    __syncthreads();
  }
#pragma unroll
  for (int mi = 0; mi < 2; ++mi)
#pragma unroll
    for (int ni = 0; ni < 4; ++ni)
      store_f32(out + (size_t)(m0 + wm * 32 + mi * 16) * 2048 + n0 + wn * 64 + ni * 16,
                2048, acc[mi][ni]);
}

// ---------- launch ----------
extern "C" void kernel_launch(void* const* d_in, const int* in_sizes, int n_in,
                              void* d_out, int out_size, void* d_ws, size_t ws_size,
                              hipStream_t stream) {
  (void)in_sizes; (void)n_in; (void)out_size; (void)ws_size;
  const float* x   = (const float*)d_in[0];
  const float* Wq  = (const float*)d_in[1];
  const float* Wk  = (const float*)d_in[2];
  const float* Wog = (const float*)d_in[3];
  const float* Wig = (const float*)d_in[4];
  const float* Wo  = (const float*)d_in[5];
  const float* gnw = (const float*)d_in[6];
  float* out = (float*)d_out;

  const size_t M = 8192, H = 2048, DK = 128;
  char* p = (char*)d_ws;
  auto take = [&](size_t bytes) { char* r = p; p += (bytes + 255) & ~(size_t)255; return r; };
  u16*   xb    = (u16*)take(M * H * 2);
  u16*   wqb   = (u16*)take(DK * H * 2);
  u16*   wkb   = (u16*)take(DK * H * 2);
  u16*   wogb  = (u16*)take(H * H * 2);
  u16*   wigb  = (u16*)take(H * H * 2);
  u16*   wob   = (u16*)take(H * H * 2);
  float* sq    = (float*)take(M * DK * 4);
  float* sk    = (float*)take(M * DK * 4);
  u16*   Ab    = (u16*)take((size_t)128 * 64 * 64 * 2);
  u16*   qgb   = (u16*)take(M * DK * 2);
  u16*   kdb   = (u16*)take(M * DK * 2);
  float* decay = (float*)take((size_t)128 * 128 * 4);
  float* o_f   = (float*)take(M * H * 4);
  u16*   o_b   = (u16*)take(M * H * 2);
  float* rn    = (float*)take(M * 4);
  u16*   gated = (u16*)take(M * H * 2);

  cvt_bf16_kernel<<<2048, 256, 0, stream>>>(x, xb, (int)(M * H));
  cvt_bf16_kernel<<<256, 256, 0, stream>>>(Wq, wqb, (int)(DK * H));
  cvt_bf16_kernel<<<256, 256, 0, stream>>>(Wk, wkb, (int)(DK * H));
  cvt_bf16_kernel<<<2048, 256, 0, stream>>>(Wog, wogb, (int)(H * H));
  cvt_bf16_kernel<<<2048, 256, 0, stream>>>(Wig, wigb, (int)(H * H));
  cvt_bf16_kernel<<<2048, 256, 0, stream>>>(Wo, wob, (int)(H * H));

  qk_proj_kernel<<<64, 256, 0, stream>>>(xb, wqb, wkb, sq, sk);
  act_kernel<<<1024, 256, 0, stream>>>(sq, sk);
  chunk_prep_kernel<<<128, 256, 0, stream>>>(sq, sk, Ab, qgb, kdb, decay);
  scan_kernel<<<64, 256, 0, stream>>>(xb, Ab, qgb, kdb, decay, o_f, o_b);
  rnorm_kernel<<<1024, 256, 0, stream>>>(o_f, rn);
  gate_kernel<<<dim3(64, 16), 256, 0, stream>>>(o_b, xb, wogb, wigb, o_f, rn, gnw, gated);
  out_proj_kernel<<<dim3(64, 16), 256, 0, stream>>>(gated, wob, out);
}
